// ShareBody_6657199308989
// MI455X (gfx1250) — matrix operand order repaired and checked by reading, not compiled, not run on hardware
//
#include <hip/hip_runtime.h>
#include <hip/hip_bf16.h>

typedef __attribute__((ext_vector_type(16))) _Float16 v16h;
typedef __attribute__((ext_vector_type(8)))  float    v8f;

// ---------------------------------------------------------------------------
// generic zero
// ---------------------------------------------------------------------------
__global__ void k_zero(float* __restrict__ p, long n) {
    long i = (long)blockIdx.x * blockDim.x + threadIdx.x;
    long stride = (long)gridDim.x * blockDim.x;
    for (; i < n; i += stride) p[i] = 0.0f;
}

// ---------------------------------------------------------------------------
// degree kernel: h2[dst*2+0] += 1 (indeg), h2[src*2+1] += 1 (outdeg)
// ---------------------------------------------------------------------------
__global__ void k_degree(const int* __restrict__ src, const int* __restrict__ dst,
                         float* __restrict__ h2, long E) {
    long i = (long)blockIdx.x * blockDim.x + threadIdx.x;
    long stride = (long)gridDim.x * blockDim.x;
    for (; i < E; i += stride) {
        unsafeAtomicAdd(&h2[(size_t)dst[i] * 2 + 0], 1.0f);
        unsafeAtomicAdd(&h2[(size_t)src[i] * 2 + 1], 1.0f);
    }
}

// ---------------------------------------------------------------------------
// edge gather-scatter: agg[dst] += h[src], feature width W
// ---------------------------------------------------------------------------
template <int W>
__global__ void k_scatter(const int* __restrict__ src, const int* __restrict__ dst,
                          const float* __restrict__ h, float* __restrict__ agg, long E) {
    long i = (long)blockIdx.x * blockDim.x + threadIdx.x;
    long stride = (long)gridDim.x * blockDim.x;
    for (; i < E; i += stride) {
        const float* hs = h + (size_t)src[i] * W;
        float* ad = agg + (size_t)dst[i] * W;
#pragma unroll
        for (int k = 0; k < W; ++k) unsafeAtomicAdd(&ad[k], hs[k]);
    }
}

// ---------------------------------------------------------------------------
// per-node linear + relu: out[n,:] = relu(W @ agg[n,:] + b),  W is [OUT, IN]
// ---------------------------------------------------------------------------
template <int IN, int OUT>
__global__ void k_linear(const float* __restrict__ agg, const float* __restrict__ W,
                         const float* __restrict__ b, float* __restrict__ out, long Ntot) {
    long n = (long)blockIdx.x * blockDim.x + threadIdx.x;
    if (n >= Ntot) return;
    float in[IN];
    const float* a = agg + (size_t)n * IN;
#pragma unroll
    for (int k = 0; k < IN; ++k) in[k] = a[k];
    float* o = out + (size_t)n * OUT;
#pragma unroll
    for (int j = 0; j < OUT; ++j) {
        float acc = b[j];
#pragma unroll
        for (int k = 0; k < IN; ++k) acc += W[j * IN + k] * in[k];
        o[j] = fmaxf(acc, 0.0f);
    }
}

// ---------------------------------------------------------------------------
// mean-pool 4-wide node features per graph -> seq columns 28..31
// ---------------------------------------------------------------------------
__global__ void k_pool(const float* __restrict__ h4, float* __restrict__ seq, int Npg) {
    __shared__ float s0[256], s1[256], s2[256], s3[256];
    const int g = blockIdx.x;
    const int t = threadIdx.x;
    const float* base = h4 + (size_t)g * Npg * 4;
    float a0 = 0.f, a1 = 0.f, a2 = 0.f, a3 = 0.f;
    for (int n = t; n < Npg; n += blockDim.x) {
        const float* p = base + (size_t)n * 4;
        a0 += p[0]; a1 += p[1]; a2 += p[2]; a3 += p[3];
    }
    s0[t] = a0; s1[t] = a1; s2[t] = a2; s3[t] = a3;
    __syncthreads();
    for (int off = 128; off > 0; off >>= 1) {
        if (t < off) { s0[t] += s0[t+off]; s1[t] += s1[t+off]; s2[t] += s2[t+off]; s3[t] += s3[t+off]; }
        __syncthreads();
    }
    if (t == 0) {
        float inv = 1.0f / (float)Npg;
        float* d = seq + (size_t)g * 32 + 28;
        d[0] = s0[0]*inv; d[1] = s1[0]*inv; d[2] = s2[0]*inv; d[3] = s3[0]*inv;
    }
}

// ---------------------------------------------------------------------------
// WMMA tile loaders (wave32; A/B use contiguous k = 16*(lane>>4) + i, same order in both)
// A 16x32 f16: lane -> m = lane&15; halves i = K[k0 + g*16 + i], g = lane>>4
//
// B 32x16 f16 from row-major W[n, k]: lane -> n = lane&15;
//                                    halves i = K[k0 + (lane>>4)*16 + i]
// C/D f32: lane -> n = lane&15; vgpr r -> m = (lane>>4)*8 + r
// ---------------------------------------------------------------------------
__device__ __forceinline__ v16h load_a_tile(const float* __restrict__ A, int lda,
                                            int row0, int k0, int lane) {
    const int m = lane & 15;
    const int g = (lane >> 4) & 1;
    const float* p = A + (size_t)(row0 + m) * lda + k0 + g * 16;
    v16h a;
#pragma unroll
    for (int i = 0; i < 8; ++i) a[i] = (_Float16)p[i];        // 2x b128
#pragma unroll
    for (int i = 0; i < 8; ++i) a[8 + i] = (_Float16)p[8 + i]; // 2x b128
    return a;
}

// unmasked: caller guarantees n0+15 < number of rows of W
__device__ __forceinline__ v16h load_bT_tile(const float* __restrict__ W, int ldw,
                                             int n0, int k0, int lane) {
    const int n = lane & 15;
    const int kb = ((lane >> 4) & 1) * 16;
    const float* p = W + (size_t)(n0 + n) * ldw + k0 + kb;
    v16h b;
#pragma unroll
    for (int i = 0; i < 16; ++i) b[i] = (_Float16)p[i];        // 4x b128
    return b;
}

// masked: rows >= nMax contribute zero columns; address clamped so the
// 16-float row load is always unconditional and contiguous (4x b128)
__device__ __forceinline__ v16h load_bT_tile_masked(const float* __restrict__ W, int ldw,
                                                    int n0, int k0, int lane, int nMax) {
    const int n = lane & 15;
    const int kb = ((lane >> 4) & 1) * 16;
    const int nG = n0 + n;
    const float mask = (nG < nMax) ? 1.0f : 0.0f;
    const int nSafe = (nG < nMax) ? nG : (nMax - 1);
    const float* p = W + (size_t)nSafe * ldw + k0 + kb;
    v16h b;
#pragma unroll
    for (int i = 0; i < 16; ++i) b[i] = (_Float16)(p[i] * mask);
    return b;
}

// ---------------------------------------------------------------------------
// FC1 via WMMA: seq[:,0:28] = relu(x[256,64] @ fc1_w^T[64,28] + fc1_b)
// 32 tiles (tm 0..15, tn 0..1), one wave per tile, K = 64 -> 2 chained WMMAs
// ---------------------------------------------------------------------------
__global__ void k_fc1_wmma(const float* __restrict__ x, const float* __restrict__ fc1_w,
                           const float* __restrict__ fc1_b, float* __restrict__ seq) {
    const int lane = threadIdx.x & 31;
    const int wave = threadIdx.x >> 5;
    const int tile = blockIdx.x * 8 + wave;       // 0..31
    const int tm = tile >> 1;                      // 0..15
    const int tn = tile & 1;                       // 0..1
    v8f c = {};
#pragma unroll
    for (int kk = 0; kk < 64; kk += 32) {
        v16h a = load_a_tile(x, 64, tm * 16, kk, lane);
        v16h b = load_bT_tile_masked(fc1_w, 64, tn * 16, kk, lane, 28);
        c = __builtin_amdgcn_wmma_f32_16x16x32_f16(false, a, false, b, (short)0, c, false, false);
    }
    const int n = lane & 15;
    const int nG = tn * 16 + n;
    if (nG < 28) {
        float bias = fc1_b[nG];
#pragma unroll
        for (int r = 0; r < 8; ++r) {
            int row = tm * 16 + ((lane >> 4) << 3) + r;
            float v = fmaxf(c[r] + bias, 0.0f);
            seq[(size_t)row * 32 + nG] = v;
        }
    }
}

// ---------------------------------------------------------------------------
// LSTM input projection via WMMA (hoisted out of the scan):
// gpre[256,1024] = seq[256,32] @ W_ih^T[32,1024] + b_ih + b_hh
// K = 32 exactly -> single WMMA per 16x16 tile; 1024 tiles, one wave each
// ---------------------------------------------------------------------------
__global__ void k_gates_wmma(const float* __restrict__ seq, const float* __restrict__ W_ih,
                             const float* __restrict__ b_ih, const float* __restrict__ b_hh,
                             float* __restrict__ gpre) {
    const int lane = threadIdx.x & 31;
    const int wave = threadIdx.x >> 5;
    const int tile = blockIdx.x * 8 + wave;        // 0..1023
    const int tm = tile >> 6;                       // 0..15
    const int tn = tile & 63;                       // 0..63
    v16h a = load_a_tile(seq, 32, tm * 16, 0, lane);
    v16h b = load_bT_tile(W_ih, 32, tn * 16, 0, lane);
    v8f c = {};
    c = __builtin_amdgcn_wmma_f32_16x16x32_f16(false, a, false, b, (short)0, c, false, false);
    const int n = lane & 15;
    const int nG = tn * 16 + n;
    const float bias = b_ih[nG] + b_hh[nG];
#pragma unroll
    for (int r = 0; r < 8; ++r) {
        int row = tm * 16 + ((lane >> 4) << 3) + r;
        gpre[(size_t)row * 1024 + nG] = c[r] + bias;
    }
}

// ---------------------------------------------------------------------------
// sequential LSTM recurrence: single 1024-thread workgroup (32 wave32s),
// h/c/gates in LDS, W_hh streamed from L2 each step.
// ---------------------------------------------------------------------------
__global__ void k_lstm(const float* __restrict__ gpre, const float* __restrict__ W_hh,
                       float* __restrict__ out, int T) {
    __shared__ __align__(16) float h_lds[256];
    __shared__ __align__(16) float c_lds[256];
    __shared__ float gate[1024];
    const int j = threadIdx.x;                     // 0..1023, owns gate j
    if (j < 256) { h_lds[j] = 0.0f; c_lds[j] = 0.0f; }
    __syncthreads();
    const float4* wrow = (const float4*)(W_hh + (size_t)j * 256);
    for (int t = 0; t < T; ++t) {
        float acc = gpre[(size_t)t * 1024 + j];
        const float4* hp = (const float4*)h_lds;
#pragma unroll 8
        for (int q = 0; q < 64; ++q) {
            float4 w = wrow[q];
            float4 h = hp[q];
            acc += w.x * h.x + w.y * h.y + w.z * h.z + w.w * h.w;
        }
        gate[j] = acc;
        __syncthreads();
        if (j < 256) {
            float ig = gate[j], fg = gate[256 + j], gg = gate[512 + j], og = gate[768 + j];
            float si = 1.0f / (1.0f + __expf(-ig));
            float sf = 1.0f / (1.0f + __expf(-fg));
            float so = 1.0f / (1.0f + __expf(-og));
            float cc = sf * c_lds[j] + si * tanhf(gg);
            float hn = so * tanhf(cc);
            c_lds[j] = cc;
            h_lds[j] = hn;
            out[(size_t)t * 256 + j] = hn;
        }
        __syncthreads();
    }
}

// ---------------------------------------------------------------------------
// launch
// ---------------------------------------------------------------------------
extern "C" void kernel_launch(void* const* d_in, const int* in_sizes, int n_in,
                              void* d_out, int out_size, void* d_ws, size_t ws_size,
                              hipStream_t stream) {
    const float* x        = (const float*)d_in[0];
    const int*   edge_src = (const int*)d_in[1];
    const int*   edge_dst = (const int*)d_in[2];
    // d_in[3] node_graph_ids: ids are n / Npg, computed arithmetically instead
    const float* fc1_w    = (const float*)d_in[4];
    const float* fc1_b    = (const float*)d_in[5];
    const float* g1_w     = (const float*)d_in[6];
    const float* g1_b     = (const float*)d_in[7];
    const float* g2_w     = (const float*)d_in[8];
    const float* g2_b     = (const float*)d_in[9];
    const float* g3_w     = (const float*)d_in[10];
    const float* g3_b     = (const float*)d_in[11];
    const float* W_ih     = (const float*)d_in[12];
    const float* W_hh     = (const float*)d_in[13];
    const float* b_ih     = (const float*)d_in[14];
    const float* b_hh     = (const float*)d_in[15];
    // w_omega / u_omega: dead code in the reference (result discarded)

    const int STATE_DIM = 64;
    const long Etot = in_sizes[1];
    const long Ntot = in_sizes[3];
    const int  Bq   = in_sizes[0] / STATE_DIM;   // 256
    const int  Npg  = (int)(Ntot / Bq);          // 4096

    float* ws   = (float*)d_ws;
    float* hbuf = ws;                                  // Ntot*12
    float* agg  = hbuf + (size_t)Ntot * 12;            // Ntot*12
    float* seq  = agg  + (size_t)Ntot * 12;            // Bq*32
    float* gpre = seq  + (size_t)Bq * 32;              // Bq*1024
    float* out  = (float*)d_out;                       // Bq*256

    const int TB = 256;
    const int edgeGrid = 16384;
    const int zeroGrid = 8192;
    const int nodeGrid = (int)((Ntot + TB - 1) / TB);

    // --- graph phase: degrees -> 3 x (scatter + linear/relu) ---
    k_zero<<<zeroGrid, TB, 0, stream>>>(hbuf, Ntot * 2);
    k_degree<<<edgeGrid, TB, 0, stream>>>(edge_src, edge_dst, hbuf, Etot);

    k_zero<<<zeroGrid, TB, 0, stream>>>(agg, Ntot * 2);
    k_scatter<2><<<edgeGrid, TB, 0, stream>>>(edge_src, edge_dst, hbuf, agg, Etot);
    k_linear<2, 12><<<nodeGrid, TB, 0, stream>>>(agg, g1_w, g1_b, hbuf, Ntot);

    k_zero<<<zeroGrid, TB, 0, stream>>>(agg, Ntot * 12);
    k_scatter<12><<<edgeGrid, TB, 0, stream>>>(edge_src, edge_dst, hbuf, agg, Etot);
    k_linear<12, 12><<<nodeGrid, TB, 0, stream>>>(agg, g2_w, g2_b, hbuf, Ntot);

    k_zero<<<zeroGrid, TB, 0, stream>>>(agg, Ntot * 12);
    k_scatter<12><<<edgeGrid, TB, 0, stream>>>(edge_src, edge_dst, hbuf, agg, Etot);
    k_linear<12, 4><<<nodeGrid, TB, 0, stream>>>(agg, g3_w, g3_b, hbuf, Ntot);

    // --- pooling into seq[:,28:32] and FC1 (WMMA) into seq[:,0:28] ---
    k_pool<<<Bq, 256, 0, stream>>>(hbuf, seq, Npg);
    k_fc1_wmma<<<4, 256, 0, stream>>>(x, fc1_w, fc1_b, seq);

    // --- hoisted LSTM input projection (WMMA) ---
    k_gates_wmma<<<128, 256, 0, stream>>>(seq, W_ih, b_ih, b_hh, gpre);

    // --- sequential recurrence ---
    k_lstm<<<1, 1024, 0, stream>>>(gpre, W_hh, out, Bq);
}
